// Gemma3AttentionModule_80376017977957
// MI455X (gfx1250) — compile-verified, matrix operand order checked by reading, not run on hardware
//
#include <hip/hip_runtime.h>
#include <hip/hip_bf16.h>
#include <math.h>

// ---------------------------------------------------------------------------
// Gemma3 sliding-window attention block for MI455X (gfx1250, wave32, WMMA).
// Pipeline: f32->bf16 convert, WMMA GEMMs (QKV) with async-to-LDS A staging,
// fused RMSNorm+RoPE, flash-style windowed attention, O-proj GEMM.
// ---------------------------------------------------------------------------

#define S_LEN   4096
#define D_MODEL 2560
#define NH      8
#define NKVH    4
#define DH      256
#define HD      2048    // NH*DH
#define KVD     1024    // NKVH*DH
#define WINDOW  1024
#define ATT_SCALE 0.0625f   // 256^-0.5
#define NEG_BIG (-1.0e30f)

typedef __attribute__((ext_vector_type(16))) __bf16 bf16x16;
typedef __attribute__((ext_vector_type(8)))  __bf16 bf16x8;
typedef __attribute__((ext_vector_type(8)))  float  f32x8;
typedef __attribute__((ext_vector_type(4)))  int    i32x4;

// gfx1250 async global->LDS copy (ASYNCcnt) with synchronous fallback.
#if defined(__gfx1250__) && __has_builtin(__builtin_amdgcn_global_load_async_to_lds_b128) && __has_builtin(__builtin_amdgcn_s_wait_asynccnt)
#define USE_ASYNC_LDS 1
#else
#define USE_ASYNC_LDS 0
#endif

__device__ __forceinline__ void copy_b128_to_lds(const __bf16* gp, __bf16* lp) {
#if USE_ASYNC_LDS
  // signature (from clang diagnostic): (v4i AS1* src, v4i AS3* dst, Imm i32 offset, Imm i32 cpol)
  __builtin_amdgcn_global_load_async_to_lds_b128(
      (__attribute__((address_space(1))) i32x4*)gp,
      (__attribute__((address_space(3))) i32x4*)lp, 0, 0);
#else
  *reinterpret_cast<bf16x8*>(lp) = *reinterpret_cast<const bf16x8*>(gp);
#endif
}

__device__ __forceinline__ void async_copies_drain() {
#if USE_ASYNC_LDS
  __builtin_amdgcn_s_wait_asynccnt(0);
#endif
}

// Load one 16x32 bf16 A/B fragment half for this lane.
// Per CDNA5 layout: a[0..7] = K(kb..kb+7), a[8..15] = K(kb+16..kb+23),
// where kb = (lane<16 ? 0 : 8) is already folded into `p` by the caller.
__device__ __forceinline__ bf16x16 load_frag(const __bf16* p) {
  bf16x8 lo = *reinterpret_cast<const bf16x8*>(p);
  bf16x8 hi = *reinterpret_cast<const bf16x8*>(p + 16);
  return __builtin_shufflevector(lo, hi, 0,1,2,3,4,5,6,7,8,9,10,11,12,13,14,15);
}

__device__ __forceinline__ f32x8 wmma_bf16(bf16x16 a, bf16x16 b, f32x8 c) {
  return __builtin_amdgcn_wmma_f32_16x16x32_bf16(false, a, false, b,
                                                 (short)0, c, false, false);
}

// ---------------------------------------------------------------------------
// f32 -> bf16 elementwise
// ---------------------------------------------------------------------------
__global__ void cvt_f32_bf16_kernel(const float* __restrict__ in,
                                    __bf16* __restrict__ out, int n) {
  for (int i = blockIdx.x * blockDim.x + threadIdx.x; i < n;
       i += gridDim.x * blockDim.x)
    out[i] = (__bf16)in[i];
}

// ---------------------------------------------------------------------------
// Transpose f32 (R x C) -> bf16 (C x R), tiled via LDS.
// ---------------------------------------------------------------------------
__global__ __launch_bounds__(256)
void transpose_cvt_kernel(const float* __restrict__ in,
                          __bf16* __restrict__ out, int R, int C) {
  __shared__ float tile[32][33];
  int bx = blockIdx.x * 32, by = blockIdx.y * 32;
  int tx = threadIdx.x, ty = threadIdx.y;   // block (32, 8)
#pragma unroll
  for (int j = 0; j < 32; j += 8) {
    int r = by + ty + j, c = bx + tx;
    tile[ty + j][tx] = (r < R && c < C) ? in[(size_t)r * C + c] : 0.f;
  }
  __syncthreads();
#pragma unroll
  for (int j = 0; j < 32; j += 8) {
    int r = by + tx;          // original row -> out column
    int c = bx + ty + j;      // original col -> out row
    if (c < C && r < R) out[(size_t)c * R + r] = (__bf16)tile[tx][ty + j];
  }
}

// ---------------------------------------------------------------------------
// WMMA GEMM: C(f32, MxN) = A(bf16, MxK row-major) * Bt(bf16, NxK row-major)^T
// Block = 8 waves covering a 16x512 strip of C; each wave owns 16x64
// (4 accumulators, one shared A fragment -> 4 WMMAs).
// A tile (16x32 = 1KB) is staged in LDS, double-buffered, filled with
// gfx1250 async global->LDS copies by waves 0-1 while all waves compute.
// grid = (N/512, M/16), block = 256.
// ---------------------------------------------------------------------------
__global__ __launch_bounds__(256)
void gemm_bf16_wmma_kernel(const __bf16* __restrict__ A,
                           const __bf16* __restrict__ Bt,
                           float* __restrict__ C, int M, int N, int K) {
  __shared__ __bf16 atile[2][16 * 32];

  const int tid  = threadIdx.x;
  const int lane = tid & 31;
  const int wave = tid >> 5;
  const int row0 = blockIdx.y * 16;
  const int col0 = blockIdx.x * 512 + wave * 64;
  const int colL = lane & 15;
  const int half = lane >> 4;
  const int kboff = half * 8;

  // async A-tile feeder: lanes 0..63 (waves 0-1) move 16 rows x 64B each step
  const int arow = tid >> 2;          // 0..15   (only meaningful for tid<64)
  const int aseg = (tid & 3) * 8;     // 0,8,16,24 elements
  const __bf16* agsrc = A + (size_t)(row0 + arow) * K + aseg;

  // B fragment pointers: 4 column tiles per wave
  const __bf16* bp[4];
#pragma unroll
  for (int j = 0; j < 4; ++j)
    bp[j] = Bt + (size_t)(col0 + j * 16 + colL) * K + kboff;

  f32x8 acc[4];
#pragma unroll
  for (int j = 0; j < 4; ++j)
#pragma unroll
    for (int r = 0; r < 8; ++r) acc[j][r] = 0.f;

  const int nsteps = K >> 5;   // K / 32

  // prologue: fill buffer 0 with chunk 0
  if (tid < 64) copy_b128_to_lds(agsrc, &atile[0][arow * 32 + aseg]);

  for (int s = 0; s < nsteps; ++s) {
    const int cur = s & 1;
    async_copies_drain();        // chunk s landed (no-op for waves 2-7 / fallback)
    __syncthreads();             // buf[cur] visible to all waves

    bf16x16 a = load_frag(&atile[cur][colL * 32 + kboff]);   // ds_load_b128 x2

    // kick off chunk s+1 into the other buffer (reads of it finished at s-1,
    // and every wave has passed the barrier above).
    if (s + 1 < nsteps && tid < 64)
      copy_b128_to_lds(agsrc + (size_t)(s + 1) * 32,
                       &atile[cur ^ 1][arow * 32 + aseg]);

    const int k0 = s * 32;
#pragma unroll
    for (int j = 0; j < 4; ++j) {
      bf16x16 b = load_frag(bp[j] + k0);
      acc[j] = wmma_bf16(a, b, acc[j]);
    }
  }

#pragma unroll
  for (int j = 0; j < 4; ++j) {
    int col = col0 + j * 16 + colL;
#pragma unroll
    for (int r = 0; r < 8; ++r)
      C[(size_t)(row0 + r + half * 8) * N + col] = acc[j][r];
  }
}

// ---------------------------------------------------------------------------
// Fused RMSNorm (per head, DH=256) + RoPE, f32 in -> bf16 out.
// grid = (S, nheads), block = 256 (one thread per dh element).
// ---------------------------------------------------------------------------
__global__ __launch_bounds__(256)
void norm_rope_kernel(const float* __restrict__ in, __bf16* __restrict__ out,
                      const float* __restrict__ nw,
                      const long long* __restrict__ pos, int nheads) {
  int s = blockIdx.x, h = blockIdx.y, d = threadIdx.x;
  int ld = nheads * DH;
  __shared__ float red[DH];
  __shared__ float xs[DH];

  float x = in[(size_t)s * ld + h * DH + d];
  red[d] = x * x;
  __syncthreads();
  for (int st = DH / 2; st > 0; st >>= 1) {
    if (d < st) red[d] += red[d + st];
    __syncthreads();
  }
  float var = red[0] * (1.0f / DH);
  float rs = rsqrtf(var + 1e-6f);
  float xn = x * rs * (1.0f + nw[d]);
  xs[d] = xn;
  __syncthreads();

  float partner = (d < 128) ? xs[d + 128] : xs[d - 128];
  float sgn = (d < 128) ? -1.f : 1.f;
  int i2 = (d & 127) * 2;
  float inv_freq = __powf(10000.f, -(float)i2 / 256.f);
  float fr = (float)pos[s] * inv_freq;
  float sn, cs;
  __sincosf(fr, &sn, &cs);
  out[(size_t)s * ld + h * DH + d] = (__bf16)(xn * cs + sgn * partner * sn);
}

// ---------------------------------------------------------------------------
// Sliding-window flash attention. One wave per (16-query tile, head).
// Q: bf16 [S][HD], K: bf16 [S][KVD], Vt: bf16 [KVD][S] (dh-major),
// out: bf16 [S][HD]. grid = (S/16, NH), block = 32.
// ---------------------------------------------------------------------------
__global__ __launch_bounds__(32)
void attn_swa_kernel(const __bf16* __restrict__ Q, const __bf16* __restrict__ K,
                     const __bf16* __restrict__ Vt,
                     const int* __restrict__ amask, __bf16* __restrict__ O) {
  __shared__ __bf16 plds[16 * 32];

  const int lane  = threadIdx.x;
  const int q0    = blockIdx.x * 16;
  const int h     = blockIdx.y;
  const int kvh   = h >> 1;                 // GQA: rep=2
  const int colL  = lane & 15;
  const int half  = lane >> 4;
  const int kboff = half * 8;

  // Preload Q fragments: 16 rows x 256 dims = 8 K-steps of 32.
  bf16x16 qf[8];
  const __bf16* qbase = Q + (size_t)(q0 + colL) * HD + h * DH + kboff;
#pragma unroll
  for (int kk = 0; kk < 8; ++kk) qf[kk] = load_frag(qbase + kk * 32);

  f32x8 o[16];
#pragma unroll
  for (int nt = 0; nt < 16; ++nt)
#pragma unroll
    for (int r = 0; r < 8; ++r) o[nt][r] = 0.f;

  float m[8], l[8];
#pragma unroll
  for (int r = 0; r < 8; ++r) { m[r] = NEG_BIG; l[r] = 0.f; }

  int start = q0 - (WINDOW - 16);
  if (start < 0) start = 0;
  start &= ~31;

  const __bf16* vbase = Vt + (size_t)kvh * DH * S_LEN;

  for (int kv0 = start; kv0 <= q0 + 15; kv0 += 32) {
    // ---- scores: two 16x16 tiles over 8 K-steps -------------------------
    int krow0 = kv0 + colL;       int kr0 = krow0 < S_LEN - 1 ? krow0 : S_LEN - 1;
    int krow1 = kv0 + 16 + colL;  int kr1 = krow1 < S_LEN - 1 ? krow1 : S_LEN - 1;
    const __bf16* kb0 = K + (size_t)kr0 * KVD + kvh * DH + kboff;
    const __bf16* kb1 = K + (size_t)kr1 * KVD + kvh * DH + kboff;

    f32x8 s0 = {0.f, 0.f, 0.f, 0.f, 0.f, 0.f, 0.f, 0.f};
    f32x8 s1 = {0.f, 0.f, 0.f, 0.f, 0.f, 0.f, 0.f, 0.f};
#pragma unroll
    for (int kk = 0; kk < 8; ++kk) {
      bf16x16 b0 = load_frag(kb0 + kk * 32);
      s0 = wmma_bf16(qf[kk], b0, s0);
      bf16x16 b1 = load_frag(kb1 + kk * 32);
      s1 = wmma_bf16(qf[kk], b1, s1);
    }

    // ---- mask + scale, chunk row-max ------------------------------------
    int am0 = amask[kr0], am1 = amask[kr1];
    bool v0[8], v1[8];
    float cm[8];
#pragma unroll
    for (int r = 0; r < 8; ++r) {
      int qrow = q0 + r + half * 8;
      v0[r] = (krow0 <= qrow) && (qrow - krow0 < WINDOW) && (am0 != 0);
      v1[r] = (krow1 <= qrow) && (qrow - krow1 < WINDOW) && (am1 != 0);
      float x0 = v0[r] ? s0[r] * ATT_SCALE : NEG_BIG;
      float x1 = v1[r] ? s1[r] * ATT_SCALE : NEG_BIG;
      s0[r] = x0; s1[r] = x1;
      cm[r] = fmaxf(x0, x1);
    }
#pragma unroll
    for (int off = 1; off < 16; off <<= 1)
#pragma unroll
      for (int r = 0; r < 8; ++r) cm[r] = fmaxf(cm[r], __shfl_xor(cm[r], off, 32));

    // ---- online softmax update ------------------------------------------
    float alpha[8], rsum[8];
#pragma unroll
    for (int r = 0; r < 8; ++r) {
      float mn = fmaxf(m[r], cm[r]);
      alpha[r] = __expf(m[r] - mn);
      m[r] = mn;
      float p0 = v0[r] ? __expf(s0[r] - mn) : 0.f;
      float p1 = v1[r] ? __expf(s1[r] - mn) : 0.f;
      s0[r] = p0; s1[r] = p1;
      rsum[r] = p0 + p1;
    }
#pragma unroll
    for (int off = 1; off < 16; off <<= 1)
#pragma unroll
      for (int r = 0; r < 8; ++r) rsum[r] += __shfl_xor(rsum[r], off, 32);
#pragma unroll
    for (int r = 0; r < 8; ++r) l[r] = l[r] * alpha[r] + rsum[r];
#pragma unroll
    for (int nt = 0; nt < 16; ++nt)
#pragma unroll
      for (int r = 0; r < 8; ++r) o[nt][r] *= alpha[r];

    // ---- stage P through LDS: C-layout -> A-layout ----------------------
#pragma unroll
    for (int r = 0; r < 8; ++r) {
      plds[(r + half * 8) * 32 + colL]      = (__bf16)s0[r];
      plds[(r + half * 8) * 32 + 16 + colL] = (__bf16)s1[r];
    }
    __syncthreads();
    bf16x16 pf = load_frag(&plds[colL * 32 + kboff]);
    __syncthreads();

    // ---- O += P @ V  (K-dim = 32 kv rows, 16 n-tiles of dh) -------------
#pragma unroll
    for (int nt = 0; nt < 16; ++nt) {
      const __bf16* vp = vbase + (size_t)(nt * 16 + colL) * S_LEN + kv0 + kboff;
      bf16x16 vf = load_frag(vp);
      o[nt] = wmma_bf16(pf, vf, o[nt]);
    }
  }

  // ---- normalize + store -------------------------------------------------
  float linv[8];
#pragma unroll
  for (int r = 0; r < 8; ++r) linv[r] = (l[r] > 0.f) ? 1.f / l[r] : 0.f;
#pragma unroll
  for (int nt = 0; nt < 16; ++nt)
#pragma unroll
    for (int r = 0; r < 8; ++r) {
      int qrow = q0 + r + half * 8;
      O[(size_t)qrow * HD + h * DH + nt * 16 + colL] = (__bf16)(o[nt][r] * linv[r]);
    }
}

// ---------------------------------------------------------------------------
// Host-side launcher.
// ---------------------------------------------------------------------------
extern "C" void kernel_launch(void* const* d_in, const int* in_sizes, int n_in,
                              void* d_out, int out_size, void* d_ws, size_t ws_size,
                              hipStream_t stream) {
  (void)in_sizes; (void)n_in; (void)out_size; (void)ws_size;

  const float*      hs    = (const float*)d_in[0];
  const int*        amask = (const int*)d_in[1];
  const long long*  pos   = (const long long*)d_in[2];
  const float*      wq    = (const float*)d_in[3];
  const float*      wk    = (const float*)d_in[4];
  const float*      wv    = (const float*)d_in[5];
  const float*      wo    = (const float*)d_in[6];
  const float*      qnw   = (const float*)d_in[7];
  const float*      knw   = (const float*)d_in[8];
  float*            outp  = (float*)d_out;

  char* w = (char*)d_ws;
  // ---- workspace layout (aliased; regions reused once producers die) -----
  const size_t off_xbf = 0;                                       // 20971520
  const size_t off_wqT = off_xbf + (size_t)S_LEN * D_MODEL * 2;   // 10485760
  const size_t off_wkT = off_wqT + (size_t)HD * D_MODEL * 2;      //  5242880
  const size_t off_wvT = off_wkT + (size_t)KVD * D_MODEL * 2;     //  5242880
  const size_t off_woT = off_wvT + (size_t)KVD * D_MODEL * 2;     // 10485760
  const size_t off_qf  = off_woT + (size_t)D_MODEL * HD * 2;      // 33554432
  const size_t off_kf  = off_qf  + (size_t)S_LEN * HD * 4;        // 16777216
  const size_t off_vf  = off_kf  + (size_t)S_LEN * KVD * 4;       // 16777216
  // aliases over dead regions:
  const size_t off_qbf = off_wqT;                 // 16.8MB over wqT+wkT+wvT (21MB)
  const size_t off_kbf = off_xbf;                 //  8.4MB over dead Xbf
  const size_t off_vbT = off_xbf + (size_t)S_LEN * KVD * 2; // next 8.4MB of Xbf
  const size_t off_att = off_qf;                  // 16.8MB over dead Qf32

  __bf16* Xbf  = (__bf16*)(w + off_xbf);
  __bf16* WqT  = (__bf16*)(w + off_wqT);
  __bf16* WkT  = (__bf16*)(w + off_wkT);
  __bf16* WvT  = (__bf16*)(w + off_wvT);
  __bf16* WoT  = (__bf16*)(w + off_woT);
  float*  Qf   = (float*) (w + off_qf);
  float*  Kf   = (float*) (w + off_kf);
  float*  Vf   = (float*) (w + off_vf);
  __bf16* Qbf  = (__bf16*)(w + off_qbf);
  __bf16* Kbf  = (__bf16*)(w + off_kbf);
  __bf16* VbT  = (__bf16*)(w + off_vbT);
  __bf16* Attn = (__bf16*)(w + off_att);

  dim3 tb(32, 8);

  // 1) convert activations / weights to bf16 (weights transposed to NxK)
  cvt_f32_bf16_kernel<<<4096, 256, 0, stream>>>(hs, Xbf, S_LEN * D_MODEL);
  transpose_cvt_kernel<<<dim3(HD / 32,  D_MODEL / 32), tb, 0, stream>>>(wq, WqT, D_MODEL, HD);
  transpose_cvt_kernel<<<dim3(KVD / 32, D_MODEL / 32), tb, 0, stream>>>(wk, WkT, D_MODEL, KVD);
  transpose_cvt_kernel<<<dim3(KVD / 32, D_MODEL / 32), tb, 0, stream>>>(wv, WvT, D_MODEL, KVD);
  transpose_cvt_kernel<<<dim3(D_MODEL / 32, HD / 32),  tb, 0, stream>>>(wo, WoT, HD, D_MODEL);

  // 2) QKV projections (WMMA, async-LDS staged A tiles)
  gemm_bf16_wmma_kernel<<<dim3(HD / 512,  S_LEN / 16), 256, 0, stream>>>(Xbf, WqT, Qf, S_LEN, HD,  D_MODEL);
  gemm_bf16_wmma_kernel<<<dim3(KVD / 512, S_LEN / 16), 256, 0, stream>>>(Xbf, WkT, Kf, S_LEN, KVD, D_MODEL);
  gemm_bf16_wmma_kernel<<<dim3(KVD / 512, S_LEN / 16), 256, 0, stream>>>(Xbf, WvT, Vf, S_LEN, KVD, D_MODEL);

  // 3) RMSNorm + RoPE -> bf16 (Qbf overwrites dead WqT/WkT/WvT; Kbf over Xbf)
  norm_rope_kernel<<<dim3(S_LEN, NH),   256, 0, stream>>>(Qf, Qbf, qnw, pos, NH);
  norm_rope_kernel<<<dim3(S_LEN, NKVH), 256, 0, stream>>>(Kf, Kbf, knw, pos, NKVH);

  // 4) V -> bf16, transposed to [dh][s] for contiguous PV B-fragments
  transpose_cvt_kernel<<<dim3(KVD / 32, S_LEN / 32), tb, 0, stream>>>(Vf, VbT, S_LEN, KVD);

  // 5) sliding-window flash attention (WMMA), output over dead Qf32
  attn_swa_kernel<<<dim3(S_LEN / 16, NH), 32, 0, stream>>>(Qbf, Kbf, VbT, amask, Attn);

  // 6) output projection (WMMA) -> fp32 d_out
  gemm_bf16_wmma_kernel<<<dim3(D_MODEL / 512, S_LEN / 16), 256, 0, stream>>>(Attn, WoT, outp, S_LEN, D_MODEL, HD);
}
